// ModifiedGIN_22153441312935
// MI455X (gfx1250) — compile-verified, matrix operand order checked by reading, not run on hardware
//
#include <hip/hip_runtime.h>

typedef unsigned short u16;
typedef __bf16 bf16;
typedef bf16 bf16x8 __attribute__((ext_vector_type(8)));
typedef bf16 bf16x16 __attribute__((ext_vector_type(16)));
typedef float v8f __attribute__((ext_vector_type(8)));

#define G_ 512
#define NPG_ 64
#define C_ 128
#define DEG_ 8
#define NNODES (G_ * NPG_)      /* 32768 */
#define EPG (NPG_ * DEG_)       /* 512 edges per group */

__device__ __forceinline__ u16 f2bf(float f) {
  unsigned u = __float_as_uint(f);
  unsigned r = u + 0x7FFFu + ((u >> 16) & 1u);   // round-to-nearest-even
  return (u16)(r >> 16);
}

// ---------------------------------------------------------------------------
// Exclusive prefix sum of per-group intra-pair counts (G=512, trivial work)
// ---------------------------------------------------------------------------
__global__ void incr_kernel(const int* __restrict__ num_nodes,
                            int* __restrict__ incr, int g) {
  if (threadIdx.x == 0 && blockIdx.x == 0) {
    int s = 0;
    for (int i = 0; i < g; ++i) {
      incr[i] = s;
      int n = num_nodes[i];
      s += n * (n - 1) / 2;
    }
  }
}

// ---------------------------------------------------------------------------
// fp32 W[l][k][n] -> bf16 Wt[l][n][k]  (transpose so WMMA B-fragments are
// two contiguous b128 loads per lane)
// ---------------------------------------------------------------------------
__global__ void wconv_kernel(const float* __restrict__ W, u16* __restrict__ Wt,
                             int K, int Nout) {
  const long total = (long)K * Nout;
  const float* Wl = W + (long)blockIdx.y * total;
  u16* Wtl = Wt + (long)blockIdx.y * total;
  for (long i = (long)blockIdx.x * blockDim.x + threadIdx.x; i < total;
       i += (long)gridDim.x * blockDim.x) {
    int k = (int)(i / Nout);
    int n = (int)(i % Nout);
    Wtl[(long)n * K + k] = f2bf(Wl[i]);
  }
}

// ---------------------------------------------------------------------------
// Per-group message passing: msgs = relu(x[src] + edge[pair_idx]),
// segment-sum into 32KB LDS accumulator, then emit Hcat = bf16([neigh | x])
// ---------------------------------------------------------------------------
__global__ __launch_bounds__(256)
void scatter_kernel(const float* __restrict__ x,
                    const float* __restrict__ edge_embeds,
                    const int* __restrict__ edge_index,
                    const int* __restrict__ cutoffs,
                    const int* __restrict__ num_nodes,
                    const int* __restrict__ incr_by,
                    u16* __restrict__ Hcat) {
  __shared__ float neigh[NPG_ * C_];   // 64 x 128 = 32KB
  const int g = blockIdx.x;
  const int tid = threadIdx.x;

  for (int i = tid; i < NPG_ * C_; i += 256) neigh[i] = 0.0f;
  __syncthreads();

  const int cut = cutoffs[g];
  const int ng  = num_nodes[g];
  const int ib  = incr_by[g];
  const int wave = tid >> 5;
  const int lane = tid & 31;

  for (int e = wave; e < EPG; e += 8) {
    const int eg = g * EPG + e;
    const int s = edge_index[2 * eg];
    const int d = edge_index[2 * eg + 1];
    const int a = min(s, d);
    const int b = max(s, d);
    const int a_in = a - cut;
    const long pidx = (long)((a_in * (2 * ng - a_in - 3)) >> 1)
                    + (b - 1 - cut) + ib;
    const int dl = d - cut;

    const float4 xs = *reinterpret_cast<const float4*>(x + (long)s * C_ + lane * 4);
    const float4 es = *reinterpret_cast<const float4*>(edge_embeds + pidx * C_ + lane * 4);
    float4 m;
    m.x = fmaxf(xs.x + es.x, 0.0f);
    m.y = fmaxf(xs.y + es.y, 0.0f);
    m.z = fmaxf(xs.z + es.z, 0.0f);
    m.w = fmaxf(xs.w + es.w, 0.0f);
    float* dst = &neigh[dl * C_ + lane * 4];
    atomicAdd(dst + 0, m.x);
    atomicAdd(dst + 1, m.y);
    atomicAdd(dst + 2, m.z);
    atomicAdd(dst + 3, m.w);
  }
  __syncthreads();

  // Hcat[node][0:128] = neigh, Hcat[node][128:256] = x  (bf16, row-major)
  for (int i = tid; i < NPG_ * 2 * C_; i += 256) {
    const int r = i >> 8;          // local node
    const int c = i & 255;         // column in [0,256)
    const long node = (long)g * NPG_ + r;
    const float v = (c < C_) ? neigh[r * C_ + c] : x[node * C_ + (c - C_)];
    Hcat[node * (2 * C_) + c] = f2bf(v);
  }
}

// ---------------------------------------------------------------------------
// bf16 WMMA GEMM: out = act(A[N x KDIM] @ Wt^T + bias)
//   A   : bf16 row-major, stride KDIM
//   Wt  : bf16 "W transposed" [NOUT x KDIM] row-major
// Block: 256 thr = 8 waves, owns 64 rows x NOUT cols.
// Wave: row-tile (wave&3), col-tile group (wave>>2), TPW tiles each.
// ---------------------------------------------------------------------------
template <int KDIM, int NOUT, bool ACT, bool OUTBF>
__global__ __launch_bounds__(256)
void gemm_kernel(const u16* __restrict__ A, const u16* __restrict__ Wt,
                 const float* __restrict__ bias, u16* __restrict__ outb,
                 float* __restrict__ outf) {
  constexpr int TPW = (NOUT / 16) / 2;   // col tiles per wave
  const int tid  = threadIdx.x;
  const int wave = tid >> 5;
  const int lane = tid & 31;
  const int half = lane >> 4;            // 0: lanes 0-15, 1: lanes 16-31
  const int lrow = lane & 15;
  const int rt = wave & 3;
  const int cg = wave >> 2;
  const int m0 = blockIdx.x * 64 + rt * 16;

  v8f acc[TPW] = {};

  // A fragment: row m0+lrow, K pairs {0..7,16..23} (half=0) / {8..15,24..31}
  const u16* aptr = A + (long)(m0 + lrow) * KDIM + half * 8;

#pragma unroll
  for (int k0 = 0; k0 < KDIM; k0 += 32) {
    bf16x8 alo = *reinterpret_cast<const bf16x8*>(aptr + k0);
    bf16x8 ahi = *reinterpret_cast<const bf16x8*>(aptr + k0 + 16);
    bf16x16 a = __builtin_shufflevector(alo, ahi, 0, 1, 2, 3, 4, 5, 6, 7, 8, 9,
                                        10, 11, 12, 13, 14, 15);
#pragma unroll
    for (int j = 0; j < TPW; ++j) {
      const int n0 = (cg * TPW + j) * 16;
      const u16* bptr = Wt + (long)(n0 + lrow) * KDIM + half * 8 + k0;
      bf16x8 blo = *reinterpret_cast<const bf16x8*>(bptr);
      bf16x8 bhi = *reinterpret_cast<const bf16x8*>(bptr + 16);
      bf16x16 b = __builtin_shufflevector(blo, bhi, 0, 1, 2, 3, 4, 5, 6, 7, 8,
                                          9, 10, 11, 12, 13, 14, 15);
      acc[j] = __builtin_amdgcn_wmma_f32_16x16x32_bf16(
          false, a, false, b, (short)0, acc[j], false, false);
    }
  }

  // Epilogue: C/D layout -> VGPR v holds row (v + half*8), lane holds col lrow
#pragma unroll
  for (int j = 0; j < TPW; ++j) {
    const int n = (cg * TPW + j) * 16 + lrow;
    const float bv = bias[n];
#pragma unroll
    for (int v = 0; v < 8; ++v) {
      const int row = m0 + half * 8 + v;
      float val = acc[j][v] + bv;
      if (ACT) val = (val >= 0.0f) ? val : 0.01f * val;
      if (OUTBF)
        outb[(long)row * NOUT + n] = f2bf(val);
      else
        outf[(long)row * NOUT + n] = val;
    }
  }
}

// ---------------------------------------------------------------------------
extern "C" void kernel_launch(void* const* d_in, const int* in_sizes, int n_in,
                              void* d_out, int out_size, void* d_ws,
                              size_t ws_size, hipStream_t stream) {
  const float* node_embeds = (const float*)d_in[0];
  const float* edge_embeds = (const float*)d_in[1];
  const int*   edge_index  = (const int*)d_in[2];
  const int*   cutoffs     = (const int*)d_in[3];
  const int*   num_nodes   = (const int*)d_in[4];
  const float* W1 = (const float*)d_in[5];
  const float* b1 = (const float*)d_in[6];
  const float* W2 = (const float*)d_in[7];
  const float* b2 = (const float*)d_in[8];
  const float* W3 = (const float*)d_in[9];
  const float* b3 = (const float*)d_in[10];

  char* ws = (char*)d_ws;
  size_t off = 0;
  auto alloc = [&](size_t bytes) -> void* {
    void* p = ws + off;
    off = (off + bytes + 255) & ~(size_t)255;
    return p;
  };

  int* incr = (int*)alloc(G_ * sizeof(int));
  u16* Wt1  = (u16*)alloc((size_t)3 * 256 * 256 * sizeof(u16));
  u16* Wt2  = (u16*)alloc((size_t)3 * 128 * 256 * sizeof(u16));
  u16* Wt3  = (u16*)alloc((size_t)3 * 128 * 128 * sizeof(u16));
  u16* Hcat = (u16*)alloc((size_t)NNODES * 256 * sizeof(u16));
  u16* Hmid = (u16*)alloc((size_t)NNODES * 256 * sizeof(u16));
  u16* H2   = (u16*)alloc((size_t)NNODES * 128 * sizeof(u16));
  float* Xb = (float*)alloc((size_t)NNODES * 128 * sizeof(float));
  (void)ws_size; (void)in_sizes; (void)n_in; (void)out_size;

  incr_kernel<<<1, 64, 0, stream>>>(num_nodes, incr, G_);
  wconv_kernel<<<dim3(256, 3), 256, 0, stream>>>(W1, Wt1, 256, 256);
  wconv_kernel<<<dim3(128, 3), 256, 0, stream>>>(W2, Wt2, 256, 128);
  wconv_kernel<<<dim3(64, 3), 256, 0, stream>>>(W3, Wt3, 128, 128);

  float* x_out = (float*)d_out;
  for (int l = 0; l < 3; ++l) {
    const float* xin = (l == 0) ? node_embeds : Xb;
    scatter_kernel<<<G_, 256, 0, stream>>>(xin, edge_embeds, edge_index,
                                           cutoffs, num_nodes, incr, Hcat);
    gemm_kernel<256, 256, true, true><<<NNODES / 64, 256, 0, stream>>>(
        Hcat, Wt1 + (size_t)l * 256 * 256, b1 + (size_t)l * 256, Hmid, nullptr);
    gemm_kernel<256, 128, true, true><<<NNODES / 64, 256, 0, stream>>>(
        Hmid, Wt2 + (size_t)l * 128 * 256, b2 + (size_t)l * 128, H2, nullptr);
    float* dst = (l == 2) ? x_out : Xb;
    gemm_kernel<128, 128, false, false><<<NNODES / 64, 256, 0, stream>>>(
        H2, Wt3 + (size_t)l * 128 * 128, b3 + (size_t)l * 128, nullptr, dst);
  }
}
